// LSTMAttentionCell_61512521613479
// MI455X (gfx1250) — matrix operand order repaired and checked by reading, not compiled, not run on hardware
//
#include <hip/hip_runtime.h>
#include <hip/hip_bf16.h>

// ---------------- problem constants ----------------
constexpr int Bb  = 64;    // batch
constexpr int Tt  = 512;   // time steps
constexpr int Ii  = 3;     // input dim
constexpr int Hh  = 400;   // lstm size
constexpr int KM  = 10;    // attention mixture components
constexpr int Ll  = 128;   // char len
constexpr int Aa  = 80;    // alphabet size
constexpr int G4H = 1600;  // 4*H gate dim

// GEMM tiling (wave32 WMMA 16x16x32 f16)
constexpr int KP1 = 512;   // padded K for LSTM1 / attention:  [w(80)|x(3)|h1(400)|pad]
constexpr int KP2 = 896;   // padded K for LSTM2/3:            [x(3)|h(400)|w(80)|h(400)|pad]
constexpr int KC1 = KP1 / 32;  // 16 k-chunks
constexpr int KC2 = KP2 / 32;  // 28 k-chunks
constexpr int NT  = G4H / 16;  // 100 n-tiles of gate columns

// persistent grid
constexpr int NWG    = 50;
constexpr int TPB    = 256;
constexpr int NTH    = NWG * TPB;   // 12800 threads
constexpr int NWAVES = NTH / 32;    // 400 waves == 4*100 tiles

// packed weight sizes (in u32 words): [ntile][kchunk][lane(32)][vgpr(8)]
constexpr long W1SZ = (long)NT * KC1 * 256;
constexpr long W2SZ = (long)NT * KC2 * 256;
constexpr long WASZ = 2L * KC1 * 256;        // Wa: 2 n-tiles (30->32 rows)

// ---------------- workspace layout (bytes, all 256-aligned) ----------------
constexpr size_t OFF_CNT  = 0;
constexpr size_t OFF_W1   = 256;
constexpr size_t OFF_W2   = OFF_W1  + (size_t)W1SZ * 4;
constexpr size_t OFF_W3   = OFF_W2  + (size_t)W2SZ * 4;
constexpr size_t OFF_WA   = OFF_W3  + (size_t)W2SZ * 4;
constexpr size_t OFF_XB1  = OFF_WA  + (size_t)WASZ * 4;          // f16 [64][512]
constexpr size_t OFF_XB2  = OFF_XB1 + (size_t)Bb * KP1 * 2;      // f16 [64][896]
constexpr size_t OFF_XB3  = OFF_XB2 + (size_t)Bb * KP2 * 2;      // f16 [64][896]
constexpr size_t OFF_G    = OFF_XB3 + (size_t)Bb * KP2 * 2;      // f32 [64][1600]
constexpr size_t OFF_AP   = OFF_G   + (size_t)Bb * G4H * 4;      // f32 [64][32]
constexpr size_t OFF_KAP  = OFF_AP  + (size_t)Bb * 32 * 4;       // f32 [2][64][16]
constexpr size_t OFF_PHI  = OFF_KAP + 2L * Bb * 16 * 4;          // f32 [64][128]
constexpr size_t OFF_C1   = OFF_PHI + (size_t)Bb * Ll * 4;       // f32 [64][400]
constexpr size_t OFF_C2   = OFF_C1  + (size_t)Bb * Hh * 4;
constexpr size_t OFF_C3   = OFF_C2  + (size_t)Bb * Hh * 4;
constexpr size_t OFF_VALS = OFF_C3  + (size_t)Bb * Hh * 4;       // f32 [64][128][80]
constexpr size_t WS_NEEDED = OFF_VALS + (size_t)Bb * Ll * Aa * 4;

// ---------------- types ----------------
typedef __attribute__((ext_vector_type(16))) _Float16 v16h;
typedef __attribute__((ext_vector_type(8)))  float    v8f;

// ---------------- small device helpers ----------------
__device__ __forceinline__ float sigf(float x)  { return 1.f / (1.f + __expf(-x)); }
__device__ __forceinline__ float softplusf(float x) { return x > 20.f ? x : log1pf(__expf(x)); }

__device__ __forceinline__ unsigned pack2h(float a, float b) {
  union { _Float16 h[2]; unsigned u; } t;
  t.h[0] = (_Float16)a; t.h[1] = (_Float16)b;
  return t.u;
}

// A fragment: 16x32 f16, lane L: M = L&15, half = L>>4
//   v0..3 -> K = half*8 + 2v, v4..7 -> K = 16 + half*8 + 2(v-4)   (ISA 7.12.2)
__device__ __forceinline__ v16h load_a_frag(const _Float16* xb, int kpad,
                                            int mtile, int kbase, int lane) {
  int row  = mtile * 16 + (lane & 15);
  int half = lane >> 4;
  const uint4* p = (const uint4*)(xb + (long)row * kpad + kbase);
  union { uint4 u[2]; v16h h; } t;
  t.u[0] = p[half];       // dwords half*4 .. half*4+3   (v0..v3)
  t.u[1] = p[2 + half];   // dwords 8+half*4 ..          (v4..v7)
  return t.h;
}

// B fragment: pre-packed [lane(32)][vgpr(8)] u32 -> 32B contiguous per lane
__device__ __forceinline__ v16h load_b_frag(const unsigned* wp, int lane) {
  const uint4* p = (const uint4*)(wp + lane * 8);
  union { uint4 u[2]; v16h h; } t;
  t.u[0] = p[0]; t.u[1] = p[1];
  return t.h;
}

// C = A(xb[64 x kpad]) x B(packed W^T) into gout[64 x gcols], f32 accum.
__device__ __forceinline__ void gemm_tiles(const _Float16* __restrict__ xb,
                                           const unsigned* __restrict__ wp,
                                           float* __restrict__ gout,
                                           int gcols, int ntiles, int kch, int kpad,
                                           int wv, int lane) {
  for (int tile = wv; tile < 4 * ntiles; tile += NWAVES) {
    int mtile = tile & 3, ntile = tile >> 2;
    const unsigned* wbase = wp + (long)ntile * kch * 256;
    v8f acc = {};
    for (int kc = 0; kc < kch; ++kc) {
      v16h a = load_a_frag(xb, kpad, mtile, kc * 32, lane);
      v16h b = load_b_frag(wbase + (long)kc * 256, lane);
      if (kc + 1 < kch)
        __builtin_prefetch(wbase + (long)(kc + 1) * 256 + lane * 8, 0, 0);
      acc = __builtin_amdgcn_wmma_f32_16x16x32_f16(false, a, false, b,
                                                   (short)0, acc, false, false);
    }
    // C layout: VGPR r -> row half*8+r, col = lane&15  (ISA 7.12.2)
    int half = lane >> 4;
    int col  = ntile * 16 + (lane & 15);
    int rb   = mtile * 16 + half * 8;
#pragma unroll
    for (int r = 0; r < 8; ++r)
      gout[(long)(rb + r) * gcols + col] = acc[r];
  }
}

// grid-wide barrier: release writes -> L2, arrive on atomic counter, acquire spin.
__device__ __forceinline__ void gsync(unsigned* cnt, unsigned* ep) {
  __syncthreads();
  if (threadIdx.x == 0) {
    __threadfence();                       // release: flush this WG's writes to L2
    unsigned target = ++(*ep) * (unsigned)NWG;
    atomicAdd(cnt, 1u);
    while (__hip_atomic_load(cnt, __ATOMIC_ACQUIRE, __HIP_MEMORY_SCOPE_AGENT) < target)
      __builtin_amdgcn_s_sleep(1);
  }
  __syncthreads();
}

// ---------------- weight accessors (concatenated + padded layouts) ----------------
// LSTM1 K layout: [w(0:80)|x(80:83)|h1(83:483)|0]; Wih1 is [1600 x 83] over cols 0..82
__device__ __forceinline__ float w_l1(const float* Wih, const float* Whh, int n, int k) {
  if (k < 83)  return Wih[n * 83 + k];
  if (k < 483) return Whh[n * 400 + (k - 83)];
  return 0.f;
}
// LSTM2/3 K layout: [x|h|w] (0:483) then recurrent h (483:883), zero pad
__device__ __forceinline__ float w_l23(const float* Wih, const float* Whh, int n, int k) {
  if (k < 483) return Wih[n * 483 + k];
  if (k < 883) return Whh[n * 400 + (k - 483)];
  return 0.f;
}
__device__ __forceinline__ float w_att(const float* Wa, int n, int k) {
  return (n < 30 && k < 483) ? Wa[n * 483 + k] : 0.f;
}

// ---------------- prep kernels ----------------
__global__ void k_zero(char* __restrict__ ws) {
  // zero: counter, xbuf1/2/3 (contiguous), AP+KAP+PHI (contiguous), C1/C2/C3 (contiguous)
  const size_t offs[4] = { OFF_CNT, OFF_XB1, OFF_AP, OFF_C1 };
  const long   cnts[4] = { 64, (long)(Bb*KP1 + 2*Bb*KP2) / 2, (Bb*32 + 2*Bb*16 + Bb*Ll),
                           3L * Bb * Hh };
  long total = cnts[0] + cnts[1] + cnts[2] + cnts[3];
  long stride = (long)gridDim.x * blockDim.x;
  for (long i = (long)blockIdx.x * blockDim.x + threadIdx.x; i < total; i += stride) {
    long j = i; int r = 0;
    while (j >= cnts[r]) { j -= cnts[r]; ++r; }
    ((unsigned*)(ws + offs[r]))[j] = 0u;
  }
}

__global__ void k_pack(const float* __restrict__ Wih1, const float* __restrict__ Whh1,
                       const float* __restrict__ Wa,
                       const float* __restrict__ Wih2, const float* __restrict__ Whh2,
                       const float* __restrict__ Wih3, const float* __restrict__ Whh3,
                       char* __restrict__ ws) {
  unsigned* P1 = (unsigned*)(ws + OFF_W1);
  unsigned* P2 = (unsigned*)(ws + OFF_W2);
  unsigned* P3 = (unsigned*)(ws + OFF_W3);
  unsigned* PA = (unsigned*)(ws + OFF_WA);
  long total  = W1SZ + 2 * W2SZ + WASZ;
  long stride = (long)gridDim.x * blockDim.x;
  for (long i = (long)blockIdx.x * blockDim.x + threadIdx.x; i < total; i += stride) {
    long j = i; unsigned* dst; int kch, which;
    if (j < W1SZ)               { dst = P1; kch = KC1; which = 1; }
    else if ((j -= W1SZ) < W2SZ){ dst = P2; kch = KC2; which = 2; }
    else if ((j -= W2SZ) < W2SZ){ dst = P3; kch = KC2; which = 3; }
    else { j -= W2SZ;             dst = PA; kch = KC1; which = 0; }
    long blk  = (long)kch * 256;
    int ntile = (int)(j / blk);
    int rem   = (int)(j - (long)ntile * blk);
    int kchunk = rem >> 8;
    int lane   = (rem >> 3) & 31;
    int v      = rem & 7;
    // B layout: lane = N col (mod 16), v0..3 -> K = half*8 + 2v, v4..7 -> K = 16 + half*8 + 2(v-4)
    int n = ntile * 16 + (lane & 15);
    int k = kchunk * 32 + (lane >> 4) * 8 + 2 * v + (v >> 2) * 8;
    float e0, e1;
    switch (which) {
      case 1:  e0 = w_l1 (Wih1, Whh1, n, k); e1 = w_l1 (Wih1, Whh1, n, k + 1); break;
      case 2:  e0 = w_l23(Wih2, Whh2, n, k); e1 = w_l23(Wih2, Whh2, n, k + 1); break;
      case 3:  e0 = w_l23(Wih3, Whh3, n, k); e1 = w_l23(Wih3, Whh3, n, k + 1); break;
      default: e0 = w_att(Wa, n, k);         e1 = w_att(Wa, n, k + 1);         break;
    }
    dst[j] = pack2h(e0, e1);
  }
}

__global__ void k_vals(const float* __restrict__ av, const int* __restrict__ len,
                       char* __restrict__ ws) {
  float* V = (float*)(ws + OFF_VALS);
  long total  = (long)Bb * Ll * Aa;
  long stride = (long)gridDim.x * blockDim.x;
  for (long i = (long)blockIdx.x * blockDim.x + threadIdx.x; i < total; i += stride) {
    int b = (int)(i / (Ll * Aa));
    int l = (int)((i / Aa) % Ll);
    V[i] = (l < len[b]) ? av[i] : 0.f;
  }
}

__global__ void k_initx(const float* __restrict__ xin, char* __restrict__ ws) {
  _Float16* XB1 = (_Float16*)(ws + OFF_XB1);
  _Float16* XB2 = (_Float16*)(ws + OFF_XB2);
  _Float16* XB3 = (_Float16*)(ws + OFF_XB3);
  int i = threadIdx.x;
  if (i < Bb * Ii) {
    int b = i / Ii, c = i % Ii;
    _Float16 v = (_Float16)xin[(long)b * Tt * Ii + c];   // x at t=0
    XB1[b * KP1 + 80 + c] = v;
    XB2[b * KP2 + c]      = v;
    XB3[b * KP2 + c]      = v;
  }
}

// ---------------- persistent scan kernel ----------------
__global__ __launch_bounds__(TPB) void rnn_attn_scan(
    const float* __restrict__ xin,
    const float* __restrict__ bih1, const float* __restrict__ bhh1,
    const float* __restrict__ ba,
    const float* __restrict__ bih2, const float* __restrict__ bhh2,
    const float* __restrict__ bih3, const float* __restrict__ bhh3,
    float* __restrict__ out, char* __restrict__ ws) {

  unsigned*  cnt = (unsigned*)(ws + OFF_CNT);
  const unsigned* W1 = (const unsigned*)(ws + OFF_W1);
  const unsigned* W2 = (const unsigned*)(ws + OFF_W2);
  const unsigned* W3 = (const unsigned*)(ws + OFF_W3);
  const unsigned* WA = (const unsigned*)(ws + OFF_WA);
  _Float16* XB1 = (_Float16*)(ws + OFF_XB1);
  _Float16* XB2 = (_Float16*)(ws + OFF_XB2);
  _Float16* XB3 = (_Float16*)(ws + OFF_XB3);
  float* G   = (float*)(ws + OFF_G);
  float* AP  = (float*)(ws + OFF_AP);
  float* KAP = (float*)(ws + OFF_KAP);
  float* PHI = (float*)(ws + OFF_PHI);
  float* C1  = (float*)(ws + OFF_C1);
  float* C2  = (float*)(ws + OFF_C2);
  float* C3  = (float*)(ws + OFF_C3);
  const float* VALS = (const float*)(ws + OFF_VALS);

  __shared__ unsigned ep;
  if (threadIdx.x == 0) ep = 0;
  __syncthreads();

  const int tid  = blockIdx.x * TPB + threadIdx.x;
  const int wv   = tid >> 5;
  const int lane = tid & 31;

  for (int t = 0; t < Tt; ++t) {
    // ---- P1: LSTM1 gates = [w|x|h1] x W1^T ----
    gemm_tiles(XB1, W1, G, G4H, NT, KC1, KP1, wv, lane);
    gsync(cnt, &ep);

    // ---- P2: LSTM1 elementwise -> h1, c1 ----
    for (int e = tid; e < Bb * Hh; e += NTH) {
      int b = e / Hh, j = e - b * Hh;
      const float* g = G + (long)b * G4H;
      float gi = g[j]        + bih1[j]        + bhh1[j];
      float gf = g[400 + j]  + bih1[400 + j]  + bhh1[400 + j];
      float gg = g[800 + j]  + bih1[800 + j]  + bhh1[800 + j];
      float go = g[1200 + j] + bih1[1200 + j] + bhh1[1200 + j];
      float c  = sigf(gf) * C1[e] + sigf(gi) * tanhf(gg);
      C1[e] = c;
      _Float16 h = (_Float16)(sigf(go) * tanhf(c));
      XB1[b * KP1 + 83 + j] = h;   // recurrent for LSTM1 (next step) + attention input
      XB2[b * KP2 + 3 + j]  = h;   // feeds LSTM2 this step
    }
    gsync(cnt, &ep);

    // ---- P3: attention pre-activations  ap = [w|x|h1] x Wa^T  (waves 0..7) ----
    gemm_tiles(XB1, WA, AP, 32, 2, KC1, KP1, wv, lane);
    gsync(cnt, &ep);

    // ---- P4: kappa update + phi (mixture of Gaussians over char positions) ----
    {
      const float* kold = KAP + (t & 1) * Bb * 16;
      float*       knew = KAP + ((t & 1) ^ 1) * Bb * 16;
      for (int e = tid; e < Bb * Ll; e += NTH) {
        int b = e >> 7, l = e & 127;
        float acc = 0.f, kwr = 0.f;
        const float* apb = AP + b * 32;
#pragma unroll
        for (int k = 0; k < KM; ++k) {
          float al = softplusf(apb[k] + ba[k]);
          float be = fmaxf(softplusf(apb[10 + k] + ba[10 + k]), 0.01f);
          float kn = kold[b * 16 + k] + softplusf(apb[20 + k] + ba[20 + k]) * 0.04f;
          if (k == l) kwr = kn;
          float d = kn - (float)l;
          acc += al * __expf(-d * d / be);
        }
        PHI[e] = acc;
        if (l < KM) knew[b * 16 + l] = kwr;
      }
    }
    gsync(cnt, &ep);

    // ---- P5: w = phi . vals   (masked weighted sum) ----
    for (int e = tid; e < Bb * Aa; e += NTH) {
      int b = e / Aa, a = e - b * Aa;
      const float* ph = PHI + b * Ll;
      const float* vv = VALS + ((long)b * Ll) * Aa + a;
      float acc = 0.f;
      for (int l = 0; l < Ll; ++l) acc += ph[l] * vv[(long)l * Aa];
      _Float16 h = (_Float16)acc;
      XB1[b * KP1 + a]        = h;   // w for next step's LSTM1 / attention
      XB2[b * KP2 + 403 + a]  = h;   // feeds LSTM2 this step
      XB3[b * KP2 + 403 + a]  = h;   // feeds LSTM3 this step
    }
    gsync(cnt, &ep);

    // ---- P6: LSTM2 gates = [x|h1|w|h2] x W2^T ----
    gemm_tiles(XB2, W2, G, G4H, NT, KC2, KP2, wv, lane);
    gsync(cnt, &ep);

    // ---- P7: LSTM2 elementwise -> h2, c2 ----
    for (int e = tid; e < Bb * Hh; e += NTH) {
      int b = e / Hh, j = e - b * Hh;
      const float* g = G + (long)b * G4H;
      float gi = g[j]        + bih2[j]        + bhh2[j];
      float gf = g[400 + j]  + bih2[400 + j]  + bhh2[400 + j];
      float gg = g[800 + j]  + bih2[800 + j]  + bhh2[800 + j];
      float go = g[1200 + j] + bih2[1200 + j] + bhh2[1200 + j];
      float c  = sigf(gf) * C2[e] + sigf(gi) * tanhf(gg);
      C2[e] = c;
      _Float16 h = (_Float16)(sigf(go) * tanhf(c));
      XB2[b * KP2 + 483 + j] = h;   // recurrent for LSTM2
      XB3[b * KP2 + 3 + j]   = h;   // feeds LSTM3 this step
    }
    gsync(cnt, &ep);

    // ---- P8: LSTM3 gates = [x|h2|w|h3] x W3^T ----
    gemm_tiles(XB3, W3, G, G4H, NT, KC2, KP2, wv, lane);
    gsync(cnt, &ep);

    // ---- P9: LSTM3 elementwise -> h3 (output), c3; stage x_{t+1} ----
    for (int e = tid; e < Bb * Hh; e += NTH) {
      int b = e / Hh, j = e - b * Hh;
      const float* g = G + (long)b * G4H;
      float gi = g[j]        + bih3[j]        + bhh3[j];
      float gf = g[400 + j]  + bih3[400 + j]  + bhh3[400 + j];
      float gg = g[800 + j]  + bih3[800 + j]  + bhh3[800 + j];
      float go = g[1200 + j] + bih3[1200 + j] + bhh3[1200 + j];
      float c  = sigf(gf) * C3[e] + sigf(gi) * tanhf(gg);
      C3[e] = c;
      float h = sigf(go) * tanhf(c);
      XB3[b * KP2 + 483 + j] = (_Float16)h;        // recurrent for LSTM3
      out[((long)b * Tt + t) * Hh + j] = h;        // kernel output [B,T,H]
    }
    if (t + 1 < Tt) {
      for (int e = tid; e < Bb * Ii; e += NTH) {
        int b = e / Ii, c = e - b * Ii;
        _Float16 v = (_Float16)xin[((long)b * Tt + (t + 1)) * Ii + c];
        XB1[b * KP1 + 80 + c] = v;
        XB2[b * KP2 + c]      = v;
        XB3[b * KP2 + c]      = v;
      }
    }
    gsync(cnt, &ep);
  }
}

// ---------------- host entry ----------------
extern "C" void kernel_launch(void* const* d_in, const int* in_sizes, int n_in,
                              void* d_out, int out_size, void* d_ws, size_t ws_size,
                              hipStream_t stream) {
  (void)in_sizes; (void)n_in; (void)out_size;
  if (ws_size < WS_NEEDED) return;  // defensive: not enough scratch

  const float* xin  = (const float*)d_in[0];
  const float* av   = (const float*)d_in[1];
  const int*   lens = (const int*)  d_in[2];
  const float* Wih1 = (const float*)d_in[3];
  const float* Whh1 = (const float*)d_in[4];
  const float* bih1 = (const float*)d_in[5];
  const float* bhh1 = (const float*)d_in[6];
  const float* Wa   = (const float*)d_in[7];
  const float* ba   = (const float*)d_in[8];
  const float* Wih2 = (const float*)d_in[9];
  const float* Whh2 = (const float*)d_in[10];
  const float* bih2 = (const float*)d_in[11];
  const float* bhh2 = (const float*)d_in[12];
  const float* Wih3 = (const float*)d_in[13];
  const float* Whh3 = (const float*)d_in[14];
  const float* bih3 = (const float*)d_in[15];
  const float* bhh3 = (const float*)d_in[16];
  char*  ws  = (char*)d_ws;
  float* out = (float*)d_out;

  k_zero <<<160, 256, 0, stream>>>(ws);
  k_pack <<<2048, 256, 0, stream>>>(Wih1, Whh1, Wa, Wih2, Whh2, Wih3, Whh3, ws);
  k_vals <<<1024, 256, 0, stream>>>(av, lens, ws);
  k_initx<<<1, 256, 0, stream>>>(xin, ws);
  rnn_attn_scan<<<NWG, TPB, 0, stream>>>(xin, bih1, bhh1, ba, bih2, bhh2,
                                         bih3, bhh3, out, ws);
}
